// NystromAttention_71038759076379
// MI455X (gfx1250) — compile-verified, matrix operand order checked by reading, not run on hardware
//
#include <hip/hip_runtime.h>

typedef __attribute__((ext_vector_type(16))) _Float16 v16h;
typedef __attribute__((ext_vector_type(8)))  float    v8f;
typedef __attribute__((ext_vector_type(2)))  float    v2f;
typedef __attribute__((ext_vector_type(4)))  unsigned uint32x4;
typedef __attribute__((ext_vector_type(8)))  int      int32x8;
typedef __attribute__((ext_vector_type(4)))  int      int32x4;

#define NG    48          // B*H
#define SEQ   4096
#define HD    64          // head dim == num landmarks
#define LD32  68          // f32 LDS row stride (bank stagger)
#define LDH   72          // f16 LDS row stride (bank stagger)
#define MAT   4096        // 64*64
#define QK_SCALE 0.35355339059327373f  // 64^-0.25

static __device__ inline v8f zero8() {
  v8f c;
#pragma unroll
  for (int i = 0; i < 8; ++i) c[i] = 0.0f;
  return c;
}

// One 16x16 tile (mi,ni) of C += A(64x64) @ B(64x64); f16 operands in LDS,
// f32 accumulate via v_wmma_f32_16x16x32_f16 (two K=32 steps).
static __device__ inline v8f mm64_f16(const _Float16* __restrict__ Ah,
                                      const _Float16* __restrict__ Bh,
                                      int mi, int ni, int lane, v8f c) {
  const int hl = lane >> 4, ln = lane & 15;
  const int row = mi * 16 + ln;
  const int col = ni * 16 + ln;
#pragma unroll
  for (int kb = 0; kb < 64; kb += 32) {
    v16h a, b;
#pragma unroll
    for (int v = 0; v < 8; ++v) {
      const int kk = kb + ((v & 4) ? 16 : 0) + hl * 8 + (v & 3) * 2;
      a[2 * v]     = Ah[row * LDH + kk];
      a[2 * v + 1] = Ah[row * LDH + kk + 1];
      b[2 * v]     = Bh[kk * LDH + col];
      b[2 * v + 1] = Bh[(kk + 1) * LDH + col];
    }
    c = __builtin_amdgcn_wmma_f32_16x16x32_f16(false, a, false, b, (short)0, c,
                                               false, false);
  }
  return c;
}

// f32 WMMA path (v_wmma_f32_16x16x4_f32), strided A/B; used for the
// numerically delicate Newton-Schulz / inverse-application matmuls.
static __device__ inline v8f mm64_f32(const float* __restrict__ A32, int lda,
                                      const float* __restrict__ B32, int ldb,
                                      int mi, int ni, int lane, v8f c) {
  const int hl = lane >> 4, ln = lane & 15;
  const int row = mi * 16 + ln;
  const int col = ni * 16 + ln;
#pragma unroll
  for (int kb = 0; kb < 64; kb += 4) {
    v2f a, b;
    a.x = A32[row * lda + kb + 2 * hl];
    a.y = A32[row * lda + kb + 2 * hl + 1];
    b.x = B32[(kb + 2 * hl) * ldb + col];
    b.y = B32[(kb + 2 * hl + 1) * ldb + col];
    c = __builtin_amdgcn_wmma_f32_16x16x4_f32(false, a, false, b, (short)0, c,
                                              false, false);
  }
  return c;
}

static __device__ inline void store_tile(float* dst, int ldc, int mi, int ni,
                                         int lane, v8f c) {
  const int hl = lane >> 4, ln = lane & 15;
#pragma unroll
  for (int r = 0; r < 8; ++r)
    dst[(mi * 16 + hl * 8 + r) * ldc + ni * 16 + ln] = c[r];
}

// dst = a*I - C   (64x64 tile write into LDS, stride LD32)
static __device__ inline void store_aI_minus(float* dst, float aa, int mi,
                                             int ni, int lane, v8f c) {
  const int hl = lane >> 4, ln = lane & 15;
#pragma unroll
  for (int r = 0; r < 8; ++r) {
    const int row = mi * 16 + hl * 8 + r, col = ni * 16 + ln;
    dst[row * LD32 + col] = (row == col ? aa : 0.0f) - c[r];
  }
}

// ---------------------------------------------------------------------------
__global__ void k_init(unsigned* colmax) {
  if (threadIdx.x == 0) *colmax = 0u;
}

// Per head: landmark pooling (scale folded in), kernel_2 = softmax(Ql@Kl^T),
// column-sum atomic max for the global Newton-Schulz scale.
__global__ void __launch_bounds__(256) k_pool_k2(
    const float* __restrict__ Q, const float* __restrict__ K,
    float* __restrict__ Qland, float* __restrict__ Kland,
    float* __restrict__ Km, unsigned* __restrict__ colmax) {
  __shared__ float Ql[64 * LD32];   // Q_land row-major
  __shared__ float KlT[64 * LD32];  // K_land transposed (B operand)
  __shared__ float K2[64 * LD32];
  const int g = blockIdx.x, tid = threadIdx.x;
  const int lane = tid & 31, wave = tid >> 5;
  const float* Qg = Q + (size_t)g * SEQ * HD;
  const float* Kg = K + (size_t)g * SEQ * HD;

  for (int idx = tid; idx < MAT; idx += 256) {
    const int l = idx >> 6, d = idx & 63;
    float sq = 0.f, sk = 0.f;
    for (int s = 0; s < 64; ++s) {
      sq += Qg[(l * 64 + s) * 64 + d];
      sk += Kg[(l * 64 + s) * 64 + d];
    }
    const float qv = sq * (QK_SCALE / 64.0f);
    const float kv = sk * (QK_SCALE / 64.0f);
    Ql[l * LD32 + d] = qv;
    KlT[d * LD32 + l] = kv;
    Qland[g * MAT + idx] = qv;
    Kland[g * MAT + idx] = kv;
  }
  __syncthreads();

  for (int t = wave; t < 16; t += 8) {  // 8 waves, 2 tiles each
    const int mi = t >> 2, ni = t & 3;
    v8f c = zero8();
    c = mm64_f32(Ql, LD32, KlT, LD32, mi, ni, lane, c);
    store_tile(K2, LD32, mi, ni, lane, c);
  }
  __syncthreads();

  if (tid < 64) {  // row softmax
    float mx = -1e30f;
    for (int j = 0; j < 64; ++j) mx = fmaxf(mx, K2[tid * LD32 + j]);
    float s = 0.f;
    for (int j = 0; j < 64; ++j) {
      const float e = __expf(K2[tid * LD32 + j] - mx);
      K2[tid * LD32 + j] = e;
      s += e;
    }
    const float inv = 1.0f / s;
    for (int j = 0; j < 64; ++j) K2[tid * LD32 + j] *= inv;
  }
  __syncthreads();

  for (int idx = tid; idx < MAT; idx += 256)
    Km[g * MAT + idx] = K2[(idx >> 6) * LD32 + (idx & 63)];
  if (tid < 64) {  // column sums -> global max (row sums are exactly 1)
    float cs = 0.f;
    for (int i = 0; i < 64; ++i) cs += K2[i * LD32 + tid];
    atomicMax(colmax, __float_as_uint(cs));
  }
}

__global__ void k_scale(const unsigned* colmax, float* scale) {
  if (threadIdx.x == 0)
    *scale = 1.0f / fmaxf(__uint_as_float(*colmax), 1e-20f);
}

// Newton-Schulz pseudo-inverse of kernel_2, all matmuls in f32 WMMA.
__global__ void __launch_bounds__(512) k_newton(
    const float* __restrict__ Km, const float* __restrict__ scale,
    float* __restrict__ INV) {
  __shared__ float Vm[64 * LD32];
  __shared__ float KV[64 * LD32];
  __shared__ float BS[64 * LD32];
  const int g = blockIdx.x, tid = threadIdx.x;
  const int lane = tid & 31, wave = tid >> 5;
  const int mi = wave >> 2, ni = wave & 3;
  const float* Kmg = Km + g * MAT;
  const float sc = *scale;

  for (int idx = tid; idx < MAT; idx += 512) {
    const int i = idx >> 6, j = idx & 63;
    Vm[j * LD32 + i] = sc * Kmg[idx];  // Vm = sc * Km^T
  }
  __syncthreads();

  for (int it = 0; it < 6; ++it) {
    v8f c = zero8();
    c = mm64_f32(Kmg, 64, Vm, LD32, mi, ni, lane, c);  // KV = Km @ Vm
    store_tile(KV, LD32, mi, ni, lane, c);
    store_aI_minus(BS, 7.0f, mi, ni, lane, c);         // BS = 7I - KV
    __syncthreads();
    v8f c2 = zero8();
    c2 = mm64_f32(KV, LD32, BS, LD32, mi, ni, lane, c2);
    __syncthreads();
    store_aI_minus(BS, 15.0f, mi, ni, lane, c2);       // BS = 15I - KV@(7I-KV)
    __syncthreads();
    v8f c3 = zero8();
    c3 = mm64_f32(KV, LD32, BS, LD32, mi, ni, lane, c3);
    __syncthreads();
    store_aI_minus(BS, 13.0f, mi, ni, lane, c3);       // BS = 13I - ...
    __syncthreads();
    v8f c4 = zero8();
    c4 = mm64_f32(Vm, LD32, BS, LD32, mi, ni, lane, c4);
    __syncthreads();
    {
      const int hl = lane >> 4, ln = lane & 15;
#pragma unroll
      for (int r = 0; r < 8; ++r)
        Vm[(mi * 16 + hl * 8 + r) * LD32 + ni * 16 + ln] = 0.25f * c4[r];
    }
    __syncthreads();
  }
  for (int idx = tid; idx < MAT; idx += 512)
    INV[g * MAT + idx] = Vm[(idx >> 6) * LD32 + (idx & 63)];
}

// K3V = softmax(Q_land @ K^T, over S) @ V  — single-pass flash-style
// streaming softmax (running max/sum, accumulator rescale), K read ONCE.
__global__ void __launch_bounds__(512) k_k3v(
    const float* __restrict__ Qland, const float* __restrict__ K,
    const float* __restrict__ V, float* __restrict__ K3V) {
  __shared__ _Float16 Qh[64 * LDH];   // A: Q_land
  __shared__ _Float16 KTh[64 * LDH];  // B: K-block transposed
  __shared__ _Float16 Vh[64 * LDH];   // B: V-block row-major
  __shared__ _Float16 Wh[64 * LDH];   // A: unnormalized exp weights
  __shared__ float Lb[64 * LD32];
  __shared__ float Mrow[64], Srow[64], Frow[64];
  const int g = blockIdx.x, tid = threadIdx.x;
  const int lane = tid & 31, wave = tid >> 5;
  const int mi = wave >> 2, ni = wave & 3;
  const int hl = lane >> 4, ln = lane & 15;
  const float* Kg = K + (size_t)g * SEQ * HD;
  const float* Vg = V + (size_t)g * SEQ * HD;

  for (int idx = tid; idx < MAT; idx += 512)
    Qh[(idx >> 6) * LDH + (idx & 63)] = (_Float16)Qland[g * MAT + idx];
  if (tid < 64) { Mrow[tid] = -1e30f; Srow[tid] = 0.f; Frow[tid] = 0.f; }
  __syncthreads();

  v8f acc = zero8();
  for (int blk = 0; blk < SEQ / 64; ++blk) {
    const float* Kb = Kg + blk * MAT;
    const float* Vb = Vg + blk * MAT;
    for (int idx = tid; idx < MAT; idx += 512) {
      const int r = idx >> 6, d = idx & 63;
      KTh[d * LDH + r] = (_Float16)(Kb[idx] * QK_SCALE);
      Vh[r * LDH + d] = (_Float16)Vb[idx];
    }
    if (blk + 1 < SEQ / 64) {  // warm L2/L0 for the next tile (prefetch_b8)
      const char* nk = (const char*)(Kb + MAT);
      const char* nv = (const char*)(Vb + MAT);
      __builtin_prefetch(nk + tid * 32, 0, 0);
      __builtin_prefetch(nv + tid * 32, 0, 0);
    }
    __syncthreads();

    v8f c = zero8();
    c = mm64_f16(Qh, KTh, mi, ni, lane, c);  // logits block
    store_tile(Lb, LD32, mi, ni, lane, c);
    __syncthreads();

    if (tid < 64) {  // online max/sum + unnormalized exp weights (one row)
      float bm = -1e30f;
      for (int j = 0; j < 64; ++j) bm = fmaxf(bm, Lb[tid * LD32 + j]);
      const float mn = fmaxf(Mrow[tid], bm);
      const float f = __expf(Mrow[tid] - mn);
      float s = Srow[tid] * f;
      for (int j = 0; j < 64; ++j) {
        const float e = __expf(Lb[tid * LD32 + j] - mn);
        Wh[tid * LDH + j] = (_Float16)e;
        s += e;
      }
      Mrow[tid] = mn;
      Srow[tid] = s;
      Frow[tid] = f;
    }
    __syncthreads();

#pragma unroll
    for (int r = 0; r < 8; ++r)  // rescale accumulator by exp(m_old - m_new)
      acc[r] *= Frow[mi * 16 + hl * 8 + r];
    acc = mm64_f16(Wh, Vh, mi, ni, lane, acc);
    __syncthreads();
  }
#pragma unroll
  for (int r = 0; r < 8; ++r) {  // normalize at store time
    const int row = mi * 16 + hl * 8 + r;
    K3V[g * MAT + row * 64 + ni * 16 + ln] = acc[r] / Srow[row];
  }
}

// W2 = INV @ K3V (f32 WMMA, tiny).
__global__ void __launch_bounds__(512) k_w2(const float* __restrict__ INV,
                                            const float* __restrict__ K3V,
                                            float* __restrict__ W2) {
  __shared__ float A32[64 * LD32];
  __shared__ float B32[64 * LD32];
  const int g = blockIdx.x, tid = threadIdx.x;
  const int lane = tid & 31, wave = tid >> 5;
  const int mi = wave >> 2, ni = wave & 3;
  for (int idx = tid; idx < MAT; idx += 512) {
    const int i = idx >> 6, j = idx & 63;
    A32[i * LD32 + j] = INV[g * MAT + idx];
    B32[i * LD32 + j] = K3V[g * MAT + idx];
  }
  __syncthreads();
  v8f c = zero8();
  c = mm64_f32(A32, LD32, B32, LD32, mi, ni, lane, c);
  store_tile(W2 + g * MAT, 64, mi, ni, lane, c);
}

// X = softmax(Q @ K_land^T) @ W2, one 64-row chunk of S per block.
// Q chunk (contiguous 16KB) staged global->LDS by the Tensor Data Mover
// (tensor_load_to_lds + s_wait_tensorcnt) where the builtin is available.
__global__ void __launch_bounds__(512) k_final(
    const float* __restrict__ Q, const float* __restrict__ Kland,
    const float* __restrict__ W2, float* __restrict__ out) {
  __shared__ _Float16 Qh[64 * LDH];    // A: Q chunk (scaled)
  __shared__ _Float16 KlTh[64 * LDH];  // B: K_land^T
  __shared__ _Float16 W2h[64 * LDH];   // B: W2 row-major
  __shared__ _Float16 Wh[64 * LDH];    // A: unnormalized exp weights
  __shared__ float Lb[64 * LD32];      // logits / TDM staging / out staging
  __shared__ float Srow[64];           // reciprocal row sums
  const int g = blockIdx.y, s0 = blockIdx.x * 64, tid = threadIdx.x;
  const int lane = tid & 31, wave = tid >> 5;
  const int mi = wave >> 2, ni = wave & 3;
  const float* Qg = Q + ((size_t)g * SEQ + s0) * HD;

#if __has_builtin(__builtin_amdgcn_tensor_load_to_lds)
  if (wave == 0) {  // TDM: DMA the 64x64 f32 Q chunk into Lb (1-D tile D#)
    const unsigned ldsOff = (unsigned)(size_t)(void*)Lb;
    const unsigned long long ga = (unsigned long long)(const void*)Qg;
    uint32x4 g0;
    g0[0] = 1u;                                   // count=1, user mode
    g0[1] = ldsOff;                               // lds_addr
    g0[2] = (unsigned)(ga & 0xFFFFFFFFu);         // global_addr[31:0]
    g0[3] = (unsigned)((ga >> 32) & 0x1FFFFFFu)   // global_addr[56:32]
            | (2u << 30);                         // type=2 (image)
    int32x8 g1;
    g1[0] = 0x20000;              // data_size=4B, no multicast/pad/iterate
    g1[1] = (int)(4096u << 16);   // tensor_dim0 = 4096 (elements)
    g1[2] = (int)(1u << 16);      // tensor_dim1 = 1
    g1[3] = (int)(4096u << 16);   // tile_dim0 = 4096
    g1[4] = 0;                    // tile_dim1/2 unused
    g1[5] = 4096;                 // tensor_dim0_stride = 4096
    g1[6] = 0;
    g1[7] = 0;
    int32x4 z4;
    z4[0] = z4[1] = z4[2] = z4[3] = 0;
    int32x8 z8;
#pragma unroll
    for (int i = 0; i < 8; ++i) z8[i] = 0;
    __builtin_amdgcn_tensor_load_to_lds(g0, g1, z4, z4, z8, 0);
    __builtin_amdgcn_s_wait_tensorcnt((short)0);
  }
  for (int idx = tid; idx < MAT; idx += 512) {  // other operands meanwhile
    const int r = idx >> 6, d = idx & 63;
    KlTh[d * LDH + r] = (_Float16)Kland[g * MAT + idx];  // transpose
    W2h[r * LDH + d] = (_Float16)W2[g * MAT + idx];
  }
  __syncthreads();
  {
    const float* Qraw = Lb;  // TDM landed here
    for (int idx = tid; idx < MAT; idx += 512)
      Qh[(idx >> 6) * LDH + (idx & 63)] = (_Float16)(Qraw[idx] * QK_SCALE);
  }
  __syncthreads();
#else
  for (int idx = tid; idx < MAT; idx += 512) {
    const int r = idx >> 6, d = idx & 63;
    Qh[r * LDH + d] = (_Float16)(Qg[idx] * QK_SCALE);
    KlTh[d * LDH + r] = (_Float16)Kland[g * MAT + idx];  // transpose
    W2h[r * LDH + d] = (_Float16)W2[g * MAT + idx];
  }
  __syncthreads();
#endif

  v8f c = zero8();
  c = mm64_f16(Qh, KlTh, mi, ni, lane, c);
  store_tile(Lb, LD32, mi, ni, lane, c);
  __syncthreads();
  if (tid < 64) {  // unnormalized exp weights; 1/sum folded into the store
    float mx = -1e30f;
    for (int j = 0; j < 64; ++j) mx = fmaxf(mx, Lb[tid * LD32 + j]);
    float s = 0.f;
    for (int j = 0; j < 64; ++j) {
      const float e = __expf(Lb[tid * LD32 + j] - mx);
      Wh[tid * LDH + j] = (_Float16)e;
      s += e;
    }
    Srow[tid] = 1.0f / s;
  }
  __syncthreads();
  v8f x = zero8();
  x = mm64_f16(Wh, W2h, mi, ni, lane, x);
  {
    const int hl = lane >> 4, ln = lane & 15;
#pragma unroll
    for (int r = 0; r < 8; ++r) {
      const int row = mi * 16 + hl * 8 + r;
      Lb[row * LD32 + ni * 16 + ln] = x[r] * Srow[row];
    }
  }
  __syncthreads();
  float* og = out + ((size_t)g * SEQ + s0) * HD;
  for (int idx = tid; idx < MAT; idx += 512)
    og[idx] = Lb[(idx >> 6) * LD32 + (idx & 63)];
}

extern "C" void kernel_launch(void* const* d_in, const int* in_sizes, int n_in,
                              void* d_out, int out_size, void* d_ws,
                              size_t ws_size, hipStream_t stream) {
  (void)in_sizes; (void)n_in; (void)out_size; (void)ws_size;
  const float* Q = (const float*)d_in[0];
  const float* K = (const float*)d_in[1];
  const float* V = (const float*)d_in[2];
  float* out = (float*)d_out;

  char* ws = (char*)d_ws;
  unsigned* colmax = (unsigned*)ws;
  float* scale = (float*)(ws + 4);
  float* base = (float*)(ws + 256);
  float* Qland = base;               // NG*4096 each
  float* Kland = Qland + NG * MAT;
  float* Km    = Kland + NG * MAT;
  float* INV   = Km + NG * MAT;
  float* K3V   = INV + NG * MAT;
  float* W2    = K3V + NG * MAT;

  k_init<<<1, 32, 0, stream>>>(colmax);
  k_pool_k2<<<NG, 256, 0, stream>>>(Q, K, Qland, Kland, Km, colmax);
  k_scale<<<1, 32, 0, stream>>>(colmax, scale);
  k_newton<<<NG, 512, 0, stream>>>(Km, scale, INV);
  k_k3v<<<NG, 512, 0, stream>>>(Qland, K, V, K3V);
  k_w2<<<NG, 512, 0, stream>>>(INV, K3V, W2);
  k_final<<<dim3(SEQ / 64, NG), 512, 0, stream>>>(Q, Kland, W2, out);
}
